// ASPGATLayer_56684978372728
// MI455X (gfx1250) — matrix operand order repaired and checked by reading, not compiled, not run on hardware
//
#include <hip/hip_runtime.h>
#include <hip/hip_bf16.h>
#include <math.h>

typedef __attribute__((ext_vector_type(16))) _Float16 v16h;
typedef __attribute__((ext_vector_type(8)))  float    v8f;

#define IN_FEAT  128
#define OUT_FEAT 128
#define KORD     3
#define LDH      136   // halves per LDS row (128 + 8 pad to skew banks)
#define TILE_M   64

// ---------------- helpers ----------------

__device__ __forceinline__ void atomAddF(float* p, float v) {
    __hip_atomic_fetch_add(p, v, __ATOMIC_RELAXED, __HIP_MEMORY_SCOPE_AGENT);
}

__device__ __forceinline__ void atomicMaxFloat(float* addr, float val) {
    // works for mixed signs when buffer initialized to -inf (0xFF800000)
    if (val >= 0.0f) atomicMax((int*)addr, __float_as_int(val));
    else             atomicMin((unsigned int*)addr, __float_as_uint(val));
}

// A fragment: 16x32 f16 tile, base -> [m=0][k=0], row stride ld halves.
// ISA 7.12.2: lanes 0-15 M=0..15 {K 0..7, 16..23}; lanes 16-31 {K 8..15, 24..31}
__device__ __forceinline__ v16h load_a(const _Float16* base, int ld, int lane) {
    const int m  = lane & 15;
    const int hh = (lane >> 4) << 3;     // 0 or 8
    const _Float16* row = base + m * ld;
    v16h a;
#pragma unroll
    for (int e = 0; e < 8; ++e) a[e]     = row[hh + e];
#pragma unroll
    for (int e = 0; e < 8; ++e) a[e + 8] = row[16 + hh + e];
    return a;
}

// B fragment: 32x16 f16 tile (KxN), base -> [k=0][n=0]; lane%16 = N col, elems = K {hh..hh+7, 16+hh..16+hh+7}
__device__ __forceinline__ v16h load_b(const _Float16* base, int ld, int lane) {
    const int n  = lane & 15;
    const int hh = (lane >> 4) << 3;
    v16h b;
#pragma unroll
    for (int e = 0; e < 8; ++e) b[e]     = base[(hh + e) * ld + n];
#pragma unroll
    for (int e = 0; e < 8; ++e) b[e + 8] = base[(16 + hh + e) * ld + n];
    return b;
}

// ---------------- kernels ----------------

// zero h/den/stats, set maxv = -inf
__global__ __launch_bounds__(256) void init_kernel(float* __restrict__ h,
                                                   float* __restrict__ den,
                                                   float* __restrict__ maxv,
                                                   float* __restrict__ stats, int N) {
    int idx = blockIdx.x * 256 + threadIdx.x;
    if (idx < N * OUT_FEAT) h[idx] = 0.0f;
    if (idx < N) { den[idx] = 0.0f; ((unsigned int*)maxv)[idx] = 0xFF800000u; }
    if (idx < 2 * OUT_FEAT) stats[idx] = 0.0f;
}

// p[n][m] = sum_o attW[m][o] * feature[n][o]   (one wave per node)
__global__ __launch_bounds__(256) void node_proj_kernel(const float* __restrict__ feature,
                                                        const float* __restrict__ attW,
                                                        float* __restrict__ p, int N) {
    const int lane = threadIdx.x & 31;
    const int n    = blockIdx.x * 8 + (threadIdx.x >> 5);
    if (n >= N) return;
    float4 f = ((const float4*)(feature + (size_t)n * IN_FEAT))[lane];
    float acc[3];
#pragma unroll
    for (int m = 0; m < 3; ++m) {
        float4 w = ((const float4*)(attW + m * OUT_FEAT))[lane];
        acc[m] = f.x * w.x + f.y * w.y + f.z * w.z + f.w * w.w;
    }
#pragma unroll
    for (int m = 0; m < 3; ++m) {
        float v = acc[m];
#pragma unroll
        for (int off = 16; off > 0; off >>= 1) v += __shfl_down(v, off, 32);
        if (lane == 0) p[n * 3 + m] = v;
    }
}

// xk[k][n][o] = sum_i feature[n][i] * linear[k][i][o]   -- WMMA f16->f32
__global__ __launch_bounds__(256) void xk_gemm_kernel(const float* __restrict__ feature,
                                                      const float* __restrict__ linear,
                                                      float* __restrict__ xk, int N) {
    __shared__ _Float16 sF[TILE_M * LDH];
    __shared__ _Float16 sW[IN_FEAT * LDH];
    const int tid = threadIdx.x, lane = tid & 31, wave = tid >> 5;
    const int m0g = blockIdx.x * TILE_M;

    for (int i = tid; i < TILE_M * IN_FEAT; i += 256) {
        int r = i >> 7, c = i & 127, gr = m0g + r;
        sF[r * LDH + c] = (_Float16)((gr < N) ? feature[(size_t)gr * IN_FEAT + c] : 0.0f);
    }
    for (int k = 0; k < KORD; ++k) {
        __syncthreads();   // covers sF on k=0, guards sW reuse otherwise
        for (int i = tid; i < IN_FEAT * OUT_FEAT; i += 256)
            sW[(i >> 7) * LDH + (i & 127)] = (_Float16)linear[k * IN_FEAT * OUT_FEAT + i];
        __syncthreads();
        for (int t = wave; t < 32; t += 8) {      // 4 mt x 8 nt tiles
            int mt = t >> 3, nt = t & 7;
            v8f c = {};
#pragma unroll
            for (int ks = 0; ks < 4; ++ks) {
                v16h a = load_a(sF + mt * 16 * LDH + ks * 32, LDH, lane);
                v16h b = load_b(sW + ks * 32 * LDH + nt * 16, LDH, lane);
                c = __builtin_amdgcn_wmma_f32_16x16x32_f16(false, a, false, b,
                                                           (short)0, c, false, false);
            }
            const int col   = nt * 16 + (lane & 15);
            const int rbase = m0g + mt * 16 + ((lane >> 4) << 3);
            float* dst = xk + ((size_t)k * N + rbase) * OUT_FEAT + col;
            if (rbase + 8 <= N) {          // fast path: full tile, branch-free stores
#pragma unroll
                for (int r = 0; r < 8; ++r) dst[(size_t)r * OUT_FEAT] = c[r];
            } else {                       // tail block only
#pragma unroll
                for (int r = 0; r < 8; ++r)
                    if (rbase + r < N) dst[(size_t)r * OUT_FEAT] = c[r];
            }
        }
    }
}

// per edge: att, invdist, segment-max into maxv[dst]
__global__ __launch_bounds__(256) void edge_att_kernel(const float* __restrict__ coords,
                                                       const int* __restrict__ src,
                                                       const int* __restrict__ dst,
                                                       const float* __restrict__ p,
                                                       float* __restrict__ att,
                                                       float* __restrict__ inv,
                                                       float* __restrict__ maxv, int E) {
    int e = blockIdx.x * 256 + threadIdx.x;
    if (e >= E) return;
    int s = src[e], d = dst[e];
    float dx = coords[s * 3 + 0] - coords[d * 3 + 0];
    float dy = coords[s * 3 + 1] - coords[d * 3 + 1];
    float dz = coords[s * 3 + 2] - coords[d * 3 + 2];
    float a  = dx * p[s * 3 + 0] + dy * p[s * 3 + 1] + dz * p[s * 3 + 2];
    att[e] = a;
    inv[e] = 1.0f / (dx * dx + dy * dy + dz * dz + 1.0f);
    atomicMaxFloat(maxv + d, a);
}

// per edge: ex = exp(att - max[dst]) (in place), denom[dst] += ex
__global__ __launch_bounds__(256) void edge_exp_kernel(const int* __restrict__ dst,
                                                       float* __restrict__ att,
                                                       const float* __restrict__ maxv,
                                                       float* __restrict__ den, int E) {
    int e = blockIdx.x * 256 + threadIdx.x;
    if (e >= E) return;
    int d = dst[e];
    float ex = __expf(att[e] - maxv[d]);
    att[e] = ex;
    atomAddF(den + d, ex);
}

// one wave per edge: h[dst] += alpha*invdist * xk[order][src]
__global__ __launch_bounds__(256) void edge_agg_kernel(const int* __restrict__ src,
                                                       const int* __restrict__ dst,
                                                       const int* __restrict__ order,
                                                       const float* __restrict__ ex,
                                                       const float* __restrict__ inv,
                                                       const float* __restrict__ den,
                                                       const float* __restrict__ xk,
                                                       float* __restrict__ h, int N, int E) {
    const int lane = threadIdx.x & 31;
    const int e    = blockIdx.x * 8 + (threadIdx.x >> 5);
    if (e >= E) return;
    int s = src[e], d = dst[e], k = order[e];
    float dn = den[d];
    float w  = ex[e] / (dn > 0.0f ? dn : 1.0f) * inv[e];
    float4 m4 = ((const float4*)(xk + ((size_t)k * N + s) * OUT_FEAT))[lane];
    float* hr = h + (size_t)d * OUT_FEAT + lane * 4;
    atomAddF(hr + 0, w * m4.x);
    atomAddF(hr + 1, w * m4.y);
    atomAddF(hr + 2, w * m4.z);
    atomAddF(hr + 3, w * m4.w);
}

// y = relu(h @ mlp_w^T + b) -> d_out; accumulate per-column sum/sumsq
__global__ __launch_bounds__(256) void mlp_kernel(const float* __restrict__ h,
                                                  const float* __restrict__ mlp_w,
                                                  const float* __restrict__ mlp_b,
                                                  float* __restrict__ y,
                                                  float* __restrict__ stats, int N) {
    __shared__ _Float16 sA[TILE_M * LDH];
    __shared__ _Float16 sW[OUT_FEAT * LDH];
    const int tid = threadIdx.x, lane = tid & 31, wave = tid >> 5;
    const int m0g = blockIdx.x * TILE_M;

    for (int i = tid; i < TILE_M * OUT_FEAT; i += 256) {
        int r = i >> 7, c = i & 127, gr = m0g + r;
        sA[r * LDH + c] = (_Float16)((gr < N) ? h[(size_t)gr * OUT_FEAT + c] : 0.0f);
    }
    for (int i = tid; i < OUT_FEAT * OUT_FEAT; i += 256) {
        int o = i >> 7, ii = i & 127;          // B[K=ii][N=o] = mlp_w[o][ii]
        sW[ii * LDH + o] = (_Float16)mlp_w[i];
    }
    __syncthreads();
    for (int t = wave; t < 32; t += 8) {
        int mt = t >> 3, nt = t & 7;
        v8f c = {};
#pragma unroll
        for (int ks = 0; ks < 4; ++ks) {
            v16h a = load_a(sA + mt * 16 * LDH + ks * 32, LDH, lane);
            v16h b = load_b(sW + ks * 32 * LDH + nt * 16, LDH, lane);
            c = __builtin_amdgcn_wmma_f32_16x16x32_f16(false, a, false, b,
                                                       (short)0, c, false, false);
        }
        const int col   = nt * 16 + (lane & 15);
        const int rbase = m0g + mt * 16 + ((lane >> 4) << 3);
        const float bias = mlp_b[col];
        float* dst = y + (size_t)rbase * OUT_FEAT + col;
        float lsum = 0.0f, lsq = 0.0f;
        if (rbase + 8 <= N) {              // fast path: full tile, branch-free
#pragma unroll
            for (int r = 0; r < 8; ++r) {
                float v = c[r] + bias;
                v = v > 0.0f ? v : 0.0f;
                dst[(size_t)r * OUT_FEAT] = v;
                lsum += v; lsq += v * v;
            }
        } else {                           // tail block only
#pragma unroll
            for (int r = 0; r < 8; ++r) {
                if (rbase + r < N) {
                    float v = c[r] + bias;
                    v = v > 0.0f ? v : 0.0f;
                    dst[(size_t)r * OUT_FEAT] = v;
                    lsum += v; lsq += v * v;
                }
            }
        }
        atomAddF(stats + col, lsum);
        atomAddF(stats + OUT_FEAT + col, lsq);
    }
}

// in-place BN over d_out using accumulated stats
__global__ __launch_bounds__(256) void bn_kernel(float* __restrict__ y,
                                                 const float* __restrict__ stats,
                                                 const float* __restrict__ gamma,
                                                 const float* __restrict__ beta,
                                                 int total, float invN) {
    int idx = blockIdx.x * 256 + threadIdx.x;
    if (idx >= total) return;
    int col = idx & (OUT_FEAT - 1);
    float mean = stats[col] * invN;
    float var  = stats[OUT_FEAT + col] * invN - mean * mean;
    float g = rsqrtf(var + 1e-5f);
    y[idx] = (y[idx] - mean) * g * gamma[col] + beta[col];
}

// ---------------- launch ----------------

extern "C" void kernel_launch(void* const* d_in, const int* in_sizes, int n_in,
                              void* d_out, int out_size, void* d_ws, size_t ws_size,
                              hipStream_t stream) {
    const float* feature = (const float*)d_in[0];
    const float* coords  = (const float*)d_in[1];
    const int*   src     = (const int*)d_in[2];
    const int*   dst     = (const int*)d_in[3];
    const int*   order   = (const int*)d_in[4];
    const float* linear  = (const float*)d_in[5];
    const float* attW    = (const float*)d_in[6];
    const float* mlp_w   = (const float*)d_in[7];
    const float* mlp_b   = (const float*)d_in[8];
    const float* gamma   = (const float*)d_in[9];
    const float* beta    = (const float*)d_in[10];
    float* out = (float*)d_out;

    const int N = in_sizes[0] / IN_FEAT;
    const int E = in_sizes[2];

    // workspace carve-up (floats, 256-aligned)
    float* ws = (float*)d_ws;
    size_t off = 0;
    auto take = [&](size_t n) { float* p = ws + off; off += (n + 255) & ~(size_t)255; return p; };
    float* p_proj = take((size_t)N * 3);
    float* att    = take((size_t)E);
    float* inv    = take((size_t)E);
    float* maxv   = take((size_t)N);
    float* den    = take((size_t)N);
    float* stats  = take(2 * OUT_FEAT);
    float* h      = take((size_t)N * OUT_FEAT);
    float* xk     = take((size_t)KORD * N * OUT_FEAT);

    const int initBlocks = (N * OUT_FEAT + 255) / 256;
    init_kernel<<<initBlocks, 256, 0, stream>>>(h, den, maxv, stats, N);

    node_proj_kernel<<<(N + 7) / 8, 256, 0, stream>>>(feature, attW, p_proj, N);

    xk_gemm_kernel<<<(N + TILE_M - 1) / TILE_M, 256, 0, stream>>>(feature, linear, xk, N);

    edge_att_kernel<<<(E + 255) / 256, 256, 0, stream>>>(coords, src, dst, p_proj,
                                                         att, inv, maxv, E);

    edge_exp_kernel<<<(E + 255) / 256, 256, 0, stream>>>(dst, att, maxv, den, E);

    edge_agg_kernel<<<(E + 7) / 8, 256, 0, stream>>>(src, dst, order, att, inv, den,
                                                     xk, h, N, E);

    mlp_kernel<<<(N + TILE_M - 1) / TILE_M, 256, 0, stream>>>(h, mlp_w, mlp_b, out, stats, N);

    bn_kernel<<<(N * OUT_FEAT + 255) / 256, 256, 0, stream>>>(out, stats, gamma, beta,
                                                              N * OUT_FEAT, 1.0f / (float)N);
}
